// GSensorResponse_38706245272026
// MI455X (gfx1250) — matrix operand order repaired and checked by reading, not compiled, not run on hardware
//
#include <hip/hip_runtime.h>

// ---------------- problem constants ----------------
#define kB   4
#define kNE  5000
#define kT   512
#define kNX  48
#define kNY  48
#define kS   (kNX * kNY)      // 2304
#define kH1  64
#define kH2  128
#define kNEP 5024             // NE padded to multiple of 32 (WMMA K step)

// workspace layout (bytes)
#define RESP_OFF 0u
#define SENS_OFF 81920u                                  // float resp[B*NE] then pad
#define EV_OFF   (SENS_OFF + (unsigned)(kS * kNEP * 2))  // 23,232,512

typedef __attribute__((ext_vector_type(8)))  float     v8f;
typedef __attribute__((ext_vector_type(8)))  _Float16  v8h;
typedef __attribute__((ext_vector_type(16))) _Float16  v16h;
typedef __attribute__((ext_vector_type(4)))  unsigned  u32x4;
typedef __attribute__((ext_vector_type(8)))  unsigned  u32x8;

// ---------------- stage 1: per-electron MLP ----------------
__global__ __launch_bounds__(256) void mlp_kernel(
    const float* __restrict__ sim, const float* __restrict__ W1,
    const float* __restrict__ b1,  const float* __restrict__ W2,
    const float* __restrict__ b2,  const float* __restrict__ W3,
    const float* __restrict__ b3,  float* __restrict__ resp)
{
    int tid = blockIdx.x * 256 + threadIdx.x;
    if (tid >= kB * kNE) return;
    float x = sim[tid * 2 + 0];
    float y = sim[tid * 2 + 1];
    float h1[kH1];
#pragma unroll
    for (int j = 0; j < kH1; ++j)
        h1[j] = fmaxf(fmaf(x, W1[j], fmaf(y, W1[kH1 + j], b1[j])), 0.0f);
    float r = b3[0];
    for (int k = 0; k < kH2; ++k) {
        float s = b2[k];
#pragma unroll
        for (int j = 0; j < kH1; ++j)
            s = fmaf(h1[j], W2[j * kH2 + k], s);
        r = fmaf(fmaxf(s, 0.0f), W3[k], r);
    }
    resp[tid] = r;
}

// ---------------- stage 2a: time gaussian, evT[t][e] (f16) ----------------
__global__ __launch_bounds__(256) void ev_kernel(
    const float* __restrict__ z, const float* __restrict__ m,
    _Float16* __restrict__ evT)
{
    int e = blockIdx.x * 256 + threadIdx.x;
    int t = blockIdx.y;
    if (e >= kNEP) return;
    float v = 0.0f;
    if (e < kNE) {
        float d = (float)t - z[e];
        // exp(-(t-z)^2 / (2*BIN_SIGMA)) * GAUSS_NORM/sqrt(BIN_SIGMA) * mask
        v = __expf(-d * d * 0.1f) * 0.17841241f * m[e];
    }
    evT[(size_t)t * kNEP + e] = (_Float16)v;
}

// ---------------- stage 2b: spatial gaussian, sensT[s][e] (f16) ----------------
__global__ __launch_bounds__(256) void sens_kernel(
    const float* __restrict__ sim, const float* __restrict__ resp,
    const float* __restrict__ el,  const float* __restrict__ sloc,
    _Float16* __restrict__ sensT)
{
    int e = blockIdx.x * 256 + threadIdx.x;
    int s = blockIdx.y;
    if (e >= kNEP) return;
    float v = 0.0f;
    if (e < kNE) {
        float es    = el[0];
        float inv2  = 0.5f / (es * es);
        float coef  = 100.0f / (es * 2.5066f);
        float dx = sim[e * 2 + 0] - sloc[s * 2 + 0];
        float dy = sim[e * 2 + 1] - sloc[s * 2 + 1];
        v = coef * __expf(-(dx * dx + dy * dy) * inv2) * resp[e];
    }
    sensT[(size_t)s * kNEP + e] = (_Float16)v;
}

// ---------------- TDM helper: 2-D tile DMA, global -> LDS ----------------
// Builds D# group0 (count/lds_addr/global_addr/type) and group1 (sizes/strides)
// per CDNA5 ISA 8.3/8.4 and issues TENSOR_LOAD_TO_LDS (TENSORcnt-tracked).
// data_size = 1 byte; tile = nrows x row_bytes; row stride = stride_bytes.
__device__ __forceinline__ void tdm_load_2d(unsigned lds_addr, const void* gptr,
                                            unsigned row_bytes, unsigned nrows,
                                            unsigned stride_bytes)
{
    unsigned long long ga = (unsigned long long)(uintptr_t)gptr;
    u32x4 g0 = { 1u,                                    // count=1 (valid user D#)
                 lds_addr,                              // lds_addr[31:0]
                 (unsigned)ga,                          // global_addr[31:0]
                 (unsigned)((ga >> 32) & 0x01FFFFFFu) | 0x80000000u }; // ga[56:32] | type=2
    u32x8 g1 = { 0u,                                    // wg_mask=0, data_size=0 (1B), no flags
                 row_bytes << 16,                       // tensor_dim0[15:0] (bits 63:48)
                 nrows << 16,                           // tensor_dim1[15:0] (bits 95:80)
                 row_bytes << 16,                       // tile_dim0 (bits 127:112)
                 nrows,                                 // tile_dim1 (bits 143:128)
                 stride_bytes,                          // tensor_dim0_stride[31:0]
                 0u, 0u };
    asm volatile("tensor_load_to_lds %0, %1" :: "s"(g0), "s"(g1) : "memory");
}

// ---------------- stage 3: WMMA contraction out[s][t] (+)= A(SxK) * B(KxT) ----------------
// A = sensT [S][NEP]: 128x64B tile staged per K-chunk by the Tensor Data Mover
//     (one tensor_load_to_lds from wave 0, TENSORcnt).
// B = evT   [T][NEP]: 64x64B tile staged by per-thread async DMA
//     (global_load_async_to_lds_b128, ASYNCcnt).
// Double buffered: chunk i+1 streams into LDS while chunk i feeds WMMA.
__global__ __launch_bounds__(256) void gemm_kernel(
    const _Float16* __restrict__ At, const _Float16* __restrict__ Bt,
    float* __restrict__ out, int accumulate)
{
    __shared__ __align__(16) _Float16 lds_a[2][128 * 32]; // [m][k] rows of 64B
    __shared__ __align__(16) _Float16 lds_b[2][64 * 32];  // [n][k] rows of 64B

    const int tid  = threadIdx.x;
    const int lane = tid & 31;
    const int wave = tid >> 5;
    const int wm   = wave & 3;   // 4 waves over M (4 * 32 = 128)
    const int wn   = wave >> 2;  // 2 waves over N (2 * 32 = 64)
    const int lh   = lane >> 4;  // half-wave selector
    const int ll   = lane & 15;

    const int m0 = blockIdx.x * 128;   // over S = 2304 (18 tiles)
    const int n0 = blockIdx.y * 64;    // over T = 512  (8 tiles)

    // B: per-thread 16B async slot (64 rows x 4 slots = 256 threads)
    const int brow = tid >> 2;
    const int bc   = (tid & 3) * 8;
    const _Float16* gb = Bt + (size_t)(n0 + brow) * kNEP + bc;
    const unsigned  lb_off = (unsigned)(brow * 32 + bc);

    // A: whole 128x64B tile via one TDM descriptor (uniform addresses)
    const _Float16* gaA = At + (size_t)m0 * kNEP;

    auto issue = [&](int buf, int k0) {
        if (wave == 0)
            tdm_load_2d((unsigned)(uintptr_t)&lds_a[buf][0], gaA + k0,
                        64u, 128u, (unsigned)(kNEP * 2));
        unsigned lb = (unsigned)(uintptr_t)&lds_b[buf][lb_off];
        asm volatile("global_load_async_to_lds_b128 %0, %1, off"
                     :: "v"(lb), "v"(gb + k0) : "memory");
    };

    v8f acc[2][2] = {};

    constexpr int NIT = kNEP / 32;   // 157 K steps
    issue(0, 0);                     // prologue: chunk 0 in flight

    for (int i = 0; i < NIT; ++i) {
        const int cur = i & 1;
        // chunk i is the only DMA still possibly in flight for this wave
        if (wave == 0)
            __builtin_amdgcn_s_wait_tensorcnt(0);          // A tile landed (TDM)
        asm volatile("s_wait_asynccnt 0x0" ::: "memory");  // own B slots landed
        __syncthreads();  // all waves' chunk-i data visible; other buffer fully consumed

        if (i + 1 < NIT)
            issue(1 - cur, (i + 1) * 32);   // stream chunk i+1 while computing chunk i

        // ---- load WMMA fragments from LDS ----
        // A 16x32 f16 layout: lanes 0-15 hold K{0..7,16..23}, lanes 16-31 K{8..15,24..31}
        v16h afrag[2], bfrag[2];
#pragma unroll
        for (int mt = 0; mt < 2; ++mt) {
            int row = wm * 32 + mt * 16 + ll;
            v8h lo = *(const v8h*)&lds_a[cur][row * 32 + lh * 8];
            v8h hi = *(const v8h*)&lds_a[cur][row * 32 + 16 + lh * 8];
            afrag[mt] = __builtin_shufflevector(lo, hi,
                0, 1, 2, 3, 4, 5, 6, 7, 8, 9, 10, 11, 12, 13, 14, 15);
        }
        // B staged as [n][k]: lane (half lh) reads K{8lh..8lh+7, 16+8lh..16+8lh+7} of col n
#pragma unroll
        for (int nt = 0; nt < 2; ++nt) {
            int row = wn * 32 + nt * 16 + ll;
            v8h lo = *(const v8h*)&lds_b[cur][row * 32 + lh * 8];
            v8h hi = *(const v8h*)&lds_b[cur][row * 32 + 16 + lh * 8];
            bfrag[nt] = __builtin_shufflevector(lo, hi,
                0, 1, 2, 3, 4, 5, 6, 7, 8, 9, 10, 11, 12, 13, 14, 15);
        }

#pragma unroll
        for (int mt = 0; mt < 2; ++mt)
#pragma unroll
            for (int nt = 0; nt < 2; ++nt)
                acc[mt][nt] = __builtin_amdgcn_wmma_f32_16x16x32_f16(
                    false, afrag[mt], false, bfrag[nt],
                    (short)0, acc[mt][nt], false, false);
    }

    // ---- epilogue: D 16x16 f32 layout: VGPR v -> M = v + 8*lh, N = ll ----
    // Batched RMW: all loads issued before any add/store (single wait clause).
    const int mb = m0 + wm * 32 + 8 * lh;
    const int nb = n0 + wn * 32 + ll;
    if (accumulate) {
        float old[2][2][8];
#pragma unroll
        for (int mt = 0; mt < 2; ++mt)
#pragma unroll
            for (int nt = 0; nt < 2; ++nt)
#pragma unroll
                for (int v = 0; v < 8; ++v)
                    old[mt][nt][v] = out[(size_t)(mb + mt * 16 + v) * kT + (nb + nt * 16)];
#pragma unroll
        for (int mt = 0; mt < 2; ++mt)
#pragma unroll
            for (int nt = 0; nt < 2; ++nt)
#pragma unroll
                for (int v = 0; v < 8; ++v)
                    out[(size_t)(mb + mt * 16 + v) * kT + (nb + nt * 16)] =
                        old[mt][nt][v] + acc[mt][nt][v];
    } else {
#pragma unroll
        for (int mt = 0; mt < 2; ++mt)
#pragma unroll
            for (int nt = 0; nt < 2; ++nt)
#pragma unroll
                for (int v = 0; v < 8; ++v)
                    out[(size_t)(mb + mt * 16 + v) * kT + (nb + nt * 16)] = acc[mt][nt][v];
    }
}

// ---------------- launcher ----------------
extern "C" void kernel_launch(void* const* d_in, const int* in_sizes, int n_in,
                              void* d_out, int out_size, void* d_ws, size_t ws_size,
                              hipStream_t stream)
{
    const float* sim  = (const float*)d_in[0];
    const float* zpos = (const float*)d_in[1];
    const float* mask = (const float*)d_in[2];
    const float* W1   = (const float*)d_in[3];
    const float* b1   = (const float*)d_in[4];
    const float* W2   = (const float*)d_in[5];
    const float* b2   = (const float*)d_in[6];
    const float* W3   = (const float*)d_in[7];
    const float* b3   = (const float*)d_in[8];
    const float* el   = (const float*)d_in[9];
    const float* sloc = (const float*)d_in[10];
    float* out = (float*)d_out;

    char* ws = (char*)d_ws;
    float*     resp  = (float*)(ws + RESP_OFF);
    _Float16*  sensT = (_Float16*)(ws + SENS_OFF);
    _Float16*  evT   = (_Float16*)(ws + EV_OFF);

    // stage 1: all batches at once
    mlp_kernel<<<(kB * kNE + 255) / 256, 256, 0, stream>>>(
        sim, W1, b1, W2, b2, W3, b3, resp);

    // stages 2+3 per batch (stream order serializes prep -> gemm -> next prep)
    for (int b = 0; b < kB; ++b) {
        ev_kernel<<<dim3((kNEP + 255) / 256, kT), 256, 0, stream>>>(
            zpos + b * kNE, mask + b * kNE, evT);
        sens_kernel<<<dim3((kNEP + 255) / 256, kS), 256, 0, stream>>>(
            sim + (size_t)b * kNE * 2, resp + b * kNE, el, sloc, sensT);
        gemm_kernel<<<dim3(kS / 128, kT / 64), 256, 0, stream>>>(
            sensT, evT, out, b > 0 ? 1 : 0);
    }
}
